// RIPointTransformer_4234837754421
// MI455X (gfx1250) — matrix operand order repaired and checked by reading, not compiled, not run on hardware
//
#include <hip/hip_runtime.h>
#include <hip/hip_bf16.h>
#include <math.h>

typedef __attribute__((ext_vector_type(16))) _Float16 v16h;
typedef __attribute__((ext_vector_type(8)))  float    v8f;

#define KNN_MAX 36

// ---------------------------------------------------------------------------
// Generic GEMM: C[M,N] = act( A[M,K] @ B[K,N] + bias[N] + addend[M,N] )
// fp32 in memory, f16 WMMA with fp32 accumulation.
//
// Block = 256 threads (8 waves), block tile 64x64, K-step 32.
//  - A tile staged in LDS as fp32 [64][32] (k-contiguous)
//  - B tile staged TRANSPOSED in LDS as fp32 [n][k] so fragment pairs are
//    contiguous 8-byte ds_load_b64 reads.
//  - Interior tiles use gfx1250 GLOBAL_LOAD_ASYNC_TO_LDS_B32 (ASYNCcnt) for
//    the fill; edge tiles use guarded sync loads (zero padding), so the WMMA
//    loop itself is branch-free and EXEC is all-1s.
//  - Each wave owns a 16x32 output slice: 2 v_wmma per K-step, A fragment
//    reused across both.
// ---------------------------------------------------------------------------
#define GBM 64
#define GBN 64
#define GBK 32

__global__ __launch_bounds__(256)
void gemm_wmma_kernel(const float* __restrict__ A, const float* __restrict__ B,
                      const float* __restrict__ bias, const float* __restrict__ addend,
                      float* __restrict__ C,
                      int M, int N, int K, int ldc, int coff, int relu)
{
    __shared__ float As[GBM * GBK];   // [m][k]
    __shared__ float Bs[GBN * GBK];   // [n][k]  (transposed)

    const int tid  = threadIdx.x;
    const int lane = tid & 31;
    const int wave = tid >> 5;
    const int wm   = wave >> 1;                 // 0..3  -> M sub-tile
    const int wn   = wave & 1;                  // 0..1  -> N sub-tile (32 wide)
    const int tileM = blockIdx.y * GBM;
    const int tileN = blockIdx.x * GBN;
    const int row  = lane & 15;
    const int half = lane >> 4;

    v8f acc0 = {0.f,0.f,0.f,0.f,0.f,0.f,0.f,0.f};
    v8f acc1 = {0.f,0.f,0.f,0.f,0.f,0.f,0.f,0.f};

    for (int kb = 0; kb < K; kb += GBK) {
        const bool intA = (tileM + GBM <= M) && (kb + GBK <= K);
        const bool intB = (tileN + GBN <= N) && (kb + GBK <= K);

        // ---- fill A tile: linear l = m*32 + k, consecutive lanes -> consecutive k
        if (intA) {
#pragma unroll
            for (int i = 0; i < 8; ++i) {
                int l = tid + 256 * i;
                int r = l >> 5, c = l & 31;
                unsigned loff = (unsigned)(unsigned long long)(&As[l]);      // LDS byte offset
                unsigned long long ga =
                    (unsigned long long)(A + (size_t)(tileM + r) * K + (kb + c));
                asm volatile("global_load_async_to_lds_b32 %0, %1, off"
                             :: "v"(loff), "v"(ga) : "memory");
            }
        } else {
#pragma unroll
            for (int i = 0; i < 8; ++i) {
                int l = tid + 256 * i;
                int r = l >> 5, c = l & 31;
                int gm = tileM + r, gk = kb + c;
                As[l] = (gm < M && gk < K) ? A[(size_t)gm * K + gk] : 0.f;
            }
        }
        // ---- fill B tile transposed: read l = k*64 + n (coalesced), store [n][k]
        if (intB) {
#pragma unroll
            for (int i = 0; i < 8; ++i) {
                int l = tid + 256 * i;
                int r = l >> 6, c = l & 63;
                unsigned loff = (unsigned)(unsigned long long)(&Bs[c * GBK + r]);
                unsigned long long ga =
                    (unsigned long long)(B + (size_t)(kb + r) * N + (tileN + c));
                asm volatile("global_load_async_to_lds_b32 %0, %1, off"
                             :: "v"(loff), "v"(ga) : "memory");
            }
        } else {
#pragma unroll
            for (int i = 0; i < 8; ++i) {
                int l = tid + 256 * i;
                int r = l >> 6, c = l & 63;
                int gk = kb + r, gn = tileN + c;
                Bs[c * GBK + r] = (gk < K && gn < N) ? B[(size_t)gk * N + gn] : 0.f;
            }
        }
        if (intA || intB)
            asm volatile("s_wait_asynccnt 0x0" ::: "memory");
        __syncthreads();

        if (kb + GBK < K)   // prefetch next A tile region (global_prefetch_b8)
            __builtin_prefetch(A + (size_t)(tileM + (tid >> 5)) * K + kb + GBK + (tid & 31), 0, 0);

        // ---- build fragments from LDS (contiguous b64 pairs, no guards) ----
        v16h a, b0, b1;
        const float* Arow = &As[(wm * 16 + row) * GBK];
#pragma unroll
        for (int v = 0; v < 8; ++v) {
            int ka = ((v < 4) ? 0 : 16) + half * 8 + (v & 3) * 2;   // ISA A 16x32 layout
            float2 f = *(const float2*)&Arow[ka];
            a[2 * v]     = (_Float16)f.x;
            a[2 * v + 1] = (_Float16)f.y;
        }
        const float* Brow0 = &Bs[(wn * 32 + row) * GBK];
        const float* Brow1 = &Bs[(wn * 32 + 16 + row) * GBK];
#pragma unroll
        for (int v = 0; v < 8; ++v) {
            int kB = ((v < 4) ? 0 : 16) + half * 8 + (v & 3) * 2;   // ISA B 32x16 layout
            float2 f0 = *(const float2*)&Brow0[kB];
            float2 f1 = *(const float2*)&Brow1[kB];
            b0[2 * v]     = (_Float16)f0.x;
            b0[2 * v + 1] = (_Float16)f0.y;
            b1[2 * v]     = (_Float16)f1.x;
            b1[2 * v + 1] = (_Float16)f1.y;
        }
        acc0 = __builtin_amdgcn_wmma_f32_16x16x32_f16(false, a, false, b0, (short)0, acc0,
                                                      false, false);
        acc1 = __builtin_amdgcn_wmma_f32_16x16x32_f16(false, a, false, b1, (short)0, acc1,
                                                      false, false);
        __syncthreads();
    }

    // ---- store: lane -> N, VGPR r -> M (ISA C 16x16 layout) ----
    const int n0 = tileN + wn * 32 + row;
    const int n1 = n0 + 16;
    const int mbase = tileM + wm * 16 + half * 8;
    if (n0 < N) {
        float bv = bias ? bias[n0] : 0.f;
#pragma unroll
        for (int r = 0; r < 8; ++r) {
            int m = mbase + r;
            if (m < M) {
                float val = acc0[r] + bv;
                if (addend) val += addend[(size_t)m * N + n0];
                if (relu)   val = fmaxf(val, 0.f);
                C[(size_t)m * ldc + coff + n0] = val;
            }
        }
    }
    if (n1 < N) {
        float bv = bias ? bias[n1] : 0.f;
#pragma unroll
        for (int r = 0; r < 8; ++r) {
            int m = mbase + r;
            if (m < M) {
                float val = acc1[r] + bv;
                if (addend) val += addend[(size_t)m * N + n1];
                if (relu)   val = fmaxf(val, 0.f);
                C[(size_t)m * ldc + coff + n1] = val;
            }
        }
    }
}

// ---------------------------------------------------------------------------
// Brute-force KNN with register insertion sort (k <= 36).
// ---------------------------------------------------------------------------
__global__ void knn_kernel(const float* __restrict__ q, int Mq,
                           const float* __restrict__ s, int Ns,
                           int k, int* __restrict__ out)
{
    int m = blockIdx.x * blockDim.x + threadIdx.x;
    if (m >= Mq) return;
    float qx = q[m * 3], qy = q[m * 3 + 1], qz = q[m * 3 + 2];
    float bd[KNN_MAX]; int bi[KNN_MAX];
    for (int i = 0; i < k; ++i) { bd[i] = 1e30f; bi[i] = 0; }
    for (int j = 0; j < Ns; ++j) {
        float dx = qx - s[j * 3], dy = qy - s[j * 3 + 1], dz = qz - s[j * 3 + 2];
        float d2 = dx * dx + dy * dy + dz * dz;
        if (d2 < bd[k - 1]) {
            int p = k - 1;
            while (p > 0 && bd[p - 1] > d2) { bd[p] = bd[p - 1]; bi[p] = bi[p - 1]; --p; }
            bd[p] = d2; bi[p] = j;
        }
    }
    for (int i = 0; i < k; ++i) out[(size_t)m * k + i] = bi[i];
}

// ---------------------------------------------------------------------------
// Furthest point sampling: single block, iterative argmax (matches jax scan).
// ---------------------------------------------------------------------------
__global__ void fps_kernel(const float* __restrict__ p, int Np, int m,
                           int* __restrict__ out, float* __restrict__ dist)
{
    __shared__ float sval[256];
    __shared__ int   sidx[256];
    int tid = threadIdx.x;
    for (int j = tid; j < Np; j += 256) dist[j] = 1e10f;
    if (tid == 0) out[0] = 0;
    __syncthreads();
    int last = 0;
    for (int i = 1; i < m; ++i) {
        float lx = p[last * 3], ly = p[last * 3 + 1], lz = p[last * 3 + 2];
        float best = -1.f; int bidx = 0;
        for (int j = tid; j < Np; j += 256) {
            float dx = p[j * 3] - lx, dy = p[j * 3 + 1] - ly, dz = p[j * 3 + 2] - lz;
            float d2 = dx * dx + dy * dy + dz * dz;
            float dj = fminf(dist[j], d2);
            dist[j] = dj;
            if (dj > best) { best = dj; bidx = j; }
        }
        sval[tid] = best; sidx[tid] = bidx;
        __syncthreads();
        for (int sh = 128; sh > 0; sh >>= 1) {
            if (tid < sh) {
                if (sval[tid + sh] > sval[tid] ||
                    (sval[tid + sh] == sval[tid] && sidx[tid + sh] < sidx[tid])) {
                    sval[tid] = sval[tid + sh]; sidx[tid] = sidx[tid + sh];
                }
            }
            __syncthreads();
        }
        last = sidx[0];
        if (tid == 0) out[i] = last;
        __syncthreads();
    }
}

// ---------------------------------------------------------------------------
__global__ void gather_rows_kernel(const float* __restrict__ src, const int* __restrict__ idx,
                                   float* __restrict__ out, int rows, int cols)
{
    size_t t = (size_t)blockIdx.x * blockDim.x + threadIdx.x;
    size_t total = (size_t)rows * cols;
    if (t >= total) return;
    int r = (int)(t / cols), c = (int)(t % cols);
    out[t] = src[(size_t)idx[r] * cols + c];
}

// ---------------------------------------------------------------------------
// Point-pair features: (angle(n,d), angle(n_r,d), angle(n,n_r), ||d||)
// ---------------------------------------------------------------------------
__device__ __forceinline__ float ppf_angle(float ax, float ay, float az,
                                           float bx, float by, float bz)
{
    float cx = ay * bz - az * by;
    float cy = az * bx - ax * bz;
    float cz = ax * by - ay * bx;
    float cn = sqrtf(cx * cx + cy * cy + cz * cz + 1e-12f);
    float dt = ax * bx + ay * by + az * bz;
    return atan2f(cn, dt);
}

__global__ void ppf_kernel(const float* __restrict__ pq, const float* __restrict__ nq,
                           const float* __restrict__ ps, const float* __restrict__ ns,
                           const int* __restrict__ gi, int M, int K, float* __restrict__ out)
{
    int t = blockIdx.x * blockDim.x + threadIdx.x;
    if (t >= M * K) return;
    int m = t / K;
    int j = gi[t];
    float dx = ps[j * 3]     - pq[m * 3];
    float dy = ps[j * 3 + 1] - pq[m * 3 + 1];
    float dz = ps[j * 3 + 2] - pq[m * 3 + 2];
    float dist = sqrtf(dx * dx + dy * dy + dz * dz + 1e-12f);
    float nx = nq[m * 3], ny = nq[m * 3 + 1], nz = nq[m * 3 + 2];
    float rx = ns[j * 3], ry = ns[j * 3 + 1], rz = ns[j * 3 + 2];
    out[(size_t)t * 4 + 0] = ppf_angle(nx, ny, nz, dx, dy, dz);
    out[(size_t)t * 4 + 1] = ppf_angle(rx, ry, rz, dx, dy, dz);
    out[(size_t)t * 4 + 2] = ppf_angle(nx, ny, nz, rx, ry, rz);
    out[(size_t)t * 4 + 3] = dist;
}

// ---------------------------------------------------------------------------
// Local attention over K neighbors; one thread per (point, head).
// ---------------------------------------------------------------------------
__global__ void attn_kernel(const float* __restrict__ q, const float* __restrict__ k,
                            const float* __restrict__ v, float* __restrict__ out,
                            int M, int K, int h, int nh)
{
    int t = blockIdx.x * blockDim.x + threadIdx.x;
    if (t >= M * nh) return;
    int m = t / nh, hd = t % nh;
    int dh = h / nh;
    const float* qp = q + (size_t)m * h + hd * dh;
    float scale = rsqrtf((float)dh);
    float sc[KNN_MAX];
    float mx = -1e30f;
    for (int kk = 0; kk < K; ++kk) {
        const float* kp = k + ((size_t)m * K + kk) * h + hd * dh;
        float s = 0.f;
        for (int d = 0; d < dh; ++d) s += qp[d] * kp[d];
        s *= scale;
        sc[kk] = s;
        mx = fmaxf(mx, s);
    }
    float sum = 0.f;
    for (int kk = 0; kk < K; ++kk) { sc[kk] = expf(sc[kk] - mx); sum += sc[kk]; }
    float inv = 1.f / sum;
    for (int d = 0; d < dh; ++d) {
        float o = 0.f;
        for (int kk = 0; kk < K; ++kk)
            o += sc[kk] * v[((size_t)m * K + kk) * h + hd * dh + d];
        out[(size_t)m * h + hd * dh + d] = o * inv;
    }
}

// ---------------------------------------------------------------------------
// LayerNorm over last dim; optional residual add (post-LN) and ReLU; one
// block per row. Safe in-place (each element owned by one thread).
// ---------------------------------------------------------------------------
__global__ void ln_kernel(const float* __restrict__ x, const float* __restrict__ g,
                          const float* __restrict__ b, const float* __restrict__ res,
                          int relu, float* __restrict__ out, int C)
{
    int m = blockIdx.x;
    const float* xr = x + (size_t)m * C;
    __shared__ float sred[256];
    int tid = threadIdx.x;
    float s = 0.f;
    for (int c = tid; c < C; c += 256) s += xr[c];
    sred[tid] = s; __syncthreads();
    for (int sh = 128; sh > 0; sh >>= 1) { if (tid < sh) sred[tid] += sred[tid + sh]; __syncthreads(); }
    float mean = sred[0] / C;
    __syncthreads();
    float s2 = 0.f;
    for (int c = tid; c < C; c += 256) { float d = xr[c] - mean; s2 += d * d; }
    sred[tid] = s2; __syncthreads();
    for (int sh = 128; sh > 0; sh >>= 1) { if (tid < sh) sred[tid] += sred[tid + sh]; __syncthreads(); }
    float inv = rsqrtf(sred[0] / C + 1e-5f);
    for (int c = tid; c < C; c += 256) {
        float y = (xr[c] - mean) * inv * g[c] + b[c];
        if (res)  y += res[(size_t)m * C + c];
        if (relu) y = fmaxf(y, 0.f);
        out[(size_t)m * C + c] = y;
    }
}

// ---------------------------------------------------------------------------
__global__ void colmean_kernel(const float* __restrict__ x, int M, int C, float* __restrict__ o)
{
    int c = blockIdx.x * blockDim.x + threadIdx.x;
    if (c >= C) return;
    float s = 0.f;
    for (int m = 0; m < M; ++m) s += x[(size_t)m * C + c];
    o[c] = s / (float)M;
}

// ---------------------------------------------------------------------------
// 3-NN inverse-distance interpolation: weights + apply (optionally accumulate
// into a strided output).
// ---------------------------------------------------------------------------
__global__ void interp_weights_kernel(const float* __restrict__ pd, int Md,
                                      const float* __restrict__ ps, int Ns,
                                      int* __restrict__ oi, float* __restrict__ ow)
{
    int m = blockIdx.x * blockDim.x + threadIdx.x;
    if (m >= Md) return;
    float qx = pd[m * 3], qy = pd[m * 3 + 1], qz = pd[m * 3 + 2];
    float b0 = 1e30f, b1 = 1e30f, b2 = 1e30f;
    int   i0 = 0, i1 = 0, i2 = 0;
    for (int j = 0; j < Ns; ++j) {
        float dx = qx - ps[j * 3], dy = qy - ps[j * 3 + 1], dz = qz - ps[j * 3 + 2];
        float d2 = dx * dx + dy * dy + dz * dz;
        if (d2 < b0)      { b2 = b1; i2 = i1; b1 = b0; i1 = i0; b0 = d2; i0 = j; }
        else if (d2 < b1) { b2 = b1; i2 = i1; b1 = d2; i1 = j; }
        else if (d2 < b2) { b2 = d2; i2 = j; }
    }
    float w0 = 1.f / (sqrtf(b0 + 1e-12f) + 1e-8f);
    float w1 = 1.f / (sqrtf(b1 + 1e-12f) + 1e-8f);
    float w2 = 1.f / (sqrtf(b2 + 1e-12f) + 1e-8f);
    float wsum = w0 + w1 + w2;
    oi[m * 3] = i0; oi[m * 3 + 1] = i1; oi[m * 3 + 2] = i2;
    ow[m * 3] = w0 / wsum; ow[m * 3 + 1] = w1 / wsum; ow[m * 3 + 2] = w2 / wsum;
}

__global__ void interp_apply_kernel(const float* __restrict__ xs, const int* __restrict__ oi,
                                    const float* __restrict__ ow, float* __restrict__ out,
                                    int M, int C, int ldo, int coff, int acc)
{
    size_t t = (size_t)blockIdx.x * blockDim.x + threadIdx.x;
    if (t >= (size_t)M * C) return;
    int m = (int)(t / C), c = (int)(t % C);
    float val = ow[m * 3]     * xs[(size_t)oi[m * 3]     * C + c]
              + ow[m * 3 + 1] * xs[(size_t)oi[m * 3 + 1] * C + c]
              + ow[m * 3 + 2] * xs[(size_t)oi[m * 3 + 2] * C + c];
    float* o = out + (size_t)m * ldo + coff + c;
    if (acc) *o += val; else *o = val;
}

// ===========================================================================
extern "C" void kernel_launch(void* const* d_in, const int* in_sizes, int n_in,
                              void* d_out, int out_size, void* d_ws, size_t ws_size,
                              hipStream_t stream)
{
    (void)in_sizes; (void)n_in; (void)out_size;
    static const int NSAMP[5] = {36, 24, 24, 24, 24};
    static const int STR[5]   = {1, 4, 4, 4, 4};
    static const int PL[5]    = {32, 64, 128, 256, 512};
    static const int NBLK[5]  = {2, 3, 4, 6, 3};
    static const int NL[5]    = {8192, 2048, 512, 128, 32};
    static const int HH[5]    = {32, 64, 128, 256, 256};   // min(C, 256)
    const int N0_ = 8192;

    const float* P0   = (const float*)d_in[0];
    const float* X0   = (const float*)d_in[1];
    const float* Nrm0 = (const float*)d_in[2];

    // ---- parameter cursor (make_params insertion order) ----
    int ci = 3;
    auto nxt = [&]() -> const float* { return (const float*)d_in[ci++]; };
    struct TPrm { const float *wq,*bq,*wk,*bk,*wv,*bv,*wp1,*bp1,*wp2,*bp2,*wo,*bo; };
    struct BPrm { TPrm t; const float *lng, *lnb; };
    auto readT = [&]() { TPrm t; t.wq=nxt(); t.bq=nxt(); t.wk=nxt(); t.bk=nxt();
        t.wv=nxt(); t.bv=nxt(); t.wp1=nxt(); t.bp1=nxt(); t.wp2=nxt(); t.bp2=nxt();
        t.wo=nxt(); t.bo=nxt(); return t; };
    auto readB = [&]() { BPrm b; b.t = readT(); b.lng = nxt(); b.lnb = nxt(); return b; };

    TPrm encT[5]; BPrm encB[5][5];
    for (int i = 0; i < 5; ++i) {
        encT[i] = readT();
        for (int j = 0; j < NBLK[i] - 1; ++j) encB[i][j] = readB();
    }
    const float *d5w1 = nxt(), *d5b1 = nxt(), *d5l1g = nxt(), *d5l1b = nxt(),
                *d5w2 = nxt(), *d5b2 = nxt();
    BPrm d5blk = readB();
    struct UPrm { const float *w1,*b1,*l1g,*l1b,*w2,*b2,*l2g,*l2b; BPrm blk; };
    UPrm ups[4];                    // dec4 (lvl3), dec3, dec2, dec1 (lvl0)
    for (int t = 0; t < 4; ++t) {
        UPrm& u = ups[t];
        u.w1 = nxt(); u.b1 = nxt(); u.l1g = nxt(); u.l1b = nxt();
        u.w2 = nxt(); u.b2 = nxt(); u.l2g = nxt(); u.l2b = nxt();
        u.blk = readB();
    }
    const float *cw[5], *cb[5], *mw[5], *mb[5];
    for (int i = 0; i < 5; ++i) { cw[i] = nxt(); cb[i] = nxt(); }
    for (int i = 0; i < 5; ++i) { mw[i] = nxt(); mb[i] = nxt(); }

    // ---- workspace bump allocator ----
    char* base = (char*)d_ws;
    size_t off = 0;
    auto allocB = [&](size_t bytes) -> void* {
        size_t a = (off + 255) & ~(size_t)255;
        off = a + bytes;
        return (void*)(base + a);
    };
    auto aF = [&](size_t n) { return (float*)allocB(n * sizeof(float)); };
    auto aI = [&](size_t n) { return (int*)allocB(n * sizeof(int)); };

    const float* pC[5]; const float* nC[5];
    float* pB[5]; float* nB[5];
    pC[0] = P0; nC[0] = Nrm0; pB[0] = nullptr; nB[0] = nullptr;
    for (int i = 1; i < 5; ++i) {
        pB[i] = aF((size_t)NL[i] * 3); nB[i] = aF((size_t)NL[i] * 3);
        pC[i] = pB[i]; nC[i] = nB[i];
    }
    float* xL[5]; float* xup[5]; int* giL[5]; float* ppfL[5];
    for (int i = 0; i < 5; ++i) {
        xL[i]   = aF((size_t)NL[i] * PL[i]);
        xup[i]  = aF((size_t)NL[i] * PL[i]);
        giL[i]  = aI((size_t)NL[i] * NSAMP[i]);
        ppfL[i] = aF((size_t)NL[i] * NSAMP[i] * 4);
    }
    const size_t MKH = (size_t)8192 * 36 * 32;   // max M*K*max(Cin,h) over all calls
    float* xg   = aF(MKH);
    float* phid = aF(MKH);
    float* pe   = aF(MKH);
    float* kbuf = aF(MKH);
    float* vbuf = aF(MKH);
    float* qb = aF((size_t)8192 * 256);
    float* ab = aF((size_t)8192 * 256);
    float* t0 = aF((size_t)8192 * 64);
    float* t1 = aF((size_t)8192 * 64);
    float* xq = aF((size_t)2048 * 512);
    int*   giTD  = aI((size_t)2048 * 24);
    float* ppfTD = aF((size_t)2048 * 24 * 4);
    float* gvec = aF(512); float* g2 = aF(512); float* bcv = aF(512);
    int*   i3 = aI((size_t)8192 * 3); float* w3 = aF((size_t)8192 * 3);
    float* fdist = aF(8192); int* fidx = aI(2048);
    if (off > ws_size) return;   // workspace insufficient

    // ---- launch helpers ----
    auto gemm = [&](const float* A, const float* B, const float* bias, const float* add,
                    float* C, int M, int N, int K, int ldc, int coff, int relu) {
        dim3 g((N + GBN - 1) / GBN, (M + GBM - 1) / GBM);
        gemm_wmma_kernel<<<g, 256, 0, stream>>>(A, B, bias, add, C, M, N, K, ldc, coff, relu);
    };
    auto gather = [&](const float* src, const int* idx, float* out, int rows, int cols) {
        size_t tot = (size_t)rows * cols;
        gather_rows_kernel<<<(unsigned)((tot + 255) / 256), 256, 0, stream>>>(src, idx, out, rows, cols);
    };
    auto knn = [&](const float* q, int Mq, const float* s, int Ns, int k, int* o) {
        knn_kernel<<<(Mq + 255) / 256, 256, 0, stream>>>(q, Mq, s, Ns, k, o);
    };
    auto ppf = [&](const float* pq, const float* nq, const float* ps, const float* ns,
                   const int* gi, int M, int K, float* o) {
        ppf_kernel<<<(M * K + 255) / 256, 256, 0, stream>>>(pq, nq, ps, ns, gi, M, K, o);
    };
    auto lnorm = [&](const float* x, const float* g, const float* b, const float* res,
                     int relu, float* out, int M, int C) {
        ln_kernel<<<M, 256, 0, stream>>>(x, g, b, res, relu, out, C);
    };
    auto lpt = [&](const TPrm& tp, const float* xsrc, const float* xnode,
                   const int* gi, const float* pp, int M, int K, int Cin, int h,
                   int Cout, float* out) {
        gemm(xnode, tp.wq, tp.bq, nullptr, qb, M, h, Cin, h, 0, 0);
        gather(xsrc, gi, xg, M * K, Cin);
        gemm(pp,   tp.wp1, tp.bp1, nullptr, phid, M * K, h, 4,   h, 0, 1);
        gemm(phid, tp.wp2, tp.bp2, nullptr, pe,   M * K, h, h,   h, 0, 0);
        gemm(xg,   tp.wk,  tp.bk,  pe,      kbuf, M * K, h, Cin, h, 0, 0);
        gemm(xg,   tp.wv,  tp.bv,  pe,      vbuf, M * K, h, Cin, h, 0, 0);
        attn_kernel<<<(M * 4 + 255) / 256, 256, 0, stream>>>(qb, kbuf, vbuf, ab, M, K, h, 4);
        gemm(ab, tp.wo, tp.bo, nullptr, out, M, Cout, h, Cout, 0, 0);
    };
    auto runblk = [&](const BPrm& bp, float* x, const int* gi, const float* pp,
                      int M, int K, int C, int h) {
        lpt(bp.t, x, x, gi, pp, M, K, C, h, C, t0);
        lnorm(t0, bp.lng, bp.lnb, x, 1, x, M, C);     // relu(LN(y) + x), in-place
    };

    // ================= encoder =================
    const float* p_prev = P0; const float* n_prev = Nrm0;
    const float* x_prev = X0; int Cin = 1;
    for (int i = 0; i < 5; ++i) {
        int Np = (i == 0) ? N0_ : NL[i - 1];
        int M = NL[i], C = PL[i], h = HH[i], K = NSAMP[i];
        if (STR[i] != 1) {
            fps_kernel<<<1, 256, 0, stream>>>(p_prev, Np, M, fidx, fdist);
            gather(p_prev, fidx, pB[i], M, 3);
            gather(n_prev, fidx, nB[i], M, 3);
            gather(x_prev, fidx, xq, M, Cin);
            knn(pC[i], M, p_prev, Np, K, giTD);
            ppf(pC[i], nC[i], p_prev, n_prev, giTD, M, K, ppfTD);
            lpt(encT[i], x_prev, xq, giTD, ppfTD, M, K, Cin, h, C, xL[i]);
            knn(pC[i], M, pC[i], M, K, giL[i]);
            ppf(pC[i], nC[i], pC[i], nC[i], giL[i], M, K, ppfL[i]);
        } else {
            // stride 1: transition KNN/PPF == self KNN/PPF, compute once
            knn(p_prev, M, p_prev, Np, K, giL[i]);
            ppf(p_prev, n_prev, p_prev, n_prev, giL[i], M, K, ppfL[i]);
            lpt(encT[i], x_prev, x_prev, giL[i], ppfL[i], M, K, Cin, h, C, xL[i]);
        }
        for (int j = 0; j < NBLK[i] - 1; ++j)
            runblk(encB[i][j], xL[i], giL[i], ppfL[i], M, K, C, h);
        p_prev = pC[i]; n_prev = nC[i]; x_prev = xL[i]; Cin = C;
    }

    // ================= decoder =================
    // dec5: xc = [x, relu(mean(x)@w2+b2)] @ w1 split into top/bottom halves of w1
    colmean_kernel<<<2, 256, 0, stream>>>(xL[4], 32, 512, gvec);
    gemm(gvec, d5w2, d5b2, nullptr, g2, 1, 512, 512, 512, 0, 1);
    gemm(g2, d5w1 + (size_t)512 * 512, d5b1, nullptr, bcv, 1, 512, 512, 512, 0, 0);
    gemm(xL[4], d5w1, bcv, nullptr, t0, 32, 512, 512, 512, 0, 0);
    lnorm(t0, d5l1g, d5l1b, nullptr, 1, xup[4], 32, 512);
    runblk(d5blk, xup[4], giL[4], ppfL[4], 32, NSAMP[4], 512, HH[4]);

    float* cur = xup[4]; int curC = 512;
    for (int t = 0; t < 4; ++t) {
        int lvl = 3 - t;
        const UPrm& u = ups[t];
        int Nn = NL[lvl + 1], Ncur = NL[lvl], cout = PL[lvl];
        gemm(cur, u.w2, u.b2, nullptr, t1, Nn, cout, curC, cout, 0, 0);
        lnorm(t1, u.l2g, u.l2b, nullptr, 1, t1, Nn, cout);                       // y2
        gemm(xL[lvl], u.w1, u.b1, nullptr, xup[lvl], Ncur, cout, cout, cout, 0, 0);
        lnorm(xup[lvl], u.l1g, u.l1b, nullptr, 1, xup[lvl], Ncur, cout);         // y1
        interp_weights_kernel<<<(Ncur + 255) / 256, 256, 0, stream>>>(
            pC[lvl], Ncur, pC[lvl + 1], Nn, i3, w3);
        interp_apply_kernel<<<(unsigned)(((size_t)Ncur * cout + 255) / 256), 256, 0, stream>>>(
            t1, i3, w3, xup[lvl], Ncur, cout, cout, 0, 1);                       // y1 += interp(y2)
        runblk(u.blk, xup[lvl], giL[lvl], ppfL[lvl], Ncur, NSAMP[lvl], cout, HH[lvl]);
        cur = xup[lvl]; curC = cout;
    }

    // ================= heads -> d_out [8192, 19] =================
    float* OUT = (float*)d_out;
    gemm(xup[0], cw[0], cb[0], nullptr, OUT, 8192, 17, PL[0], 19, 0, 0);
    for (int i = 1; i < 5; ++i) {
        gemm(xup[i], cw[i], cb[i], nullptr, t1, NL[i], 17, PL[i], 17, 0, 0);
        interp_weights_kernel<<<(8192 + 255) / 256, 256, 0, stream>>>(P0, 8192, pC[i], NL[i], i3, w3);
        interp_apply_kernel<<<(unsigned)(((size_t)8192 * 17 + 255) / 256), 256, 0, stream>>>(
            t1, i3, w3, OUT, 8192, 17, 19, 0, 1);
    }
    gemm(xup[0], mw[0], mb[0], nullptr, OUT, 8192, 2, PL[0], 19, 17, 0);
    for (int i = 1; i < 5; ++i) {
        gemm(xup[i], mw[i], mb[i], nullptr, t1, NL[i], 2, PL[i], 2, 0, 0);
        interp_weights_kernel<<<(8192 + 255) / 256, 256, 0, stream>>>(P0, 8192, pC[i], NL[i], i3, w3);
        interp_apply_kernel<<<(unsigned)(((size_t)8192 * 2 + 255) / 256), 256, 0, stream>>>(
            t1, i3, w3, OUT, 8192, 2, 19, 17, 1);
    }
}
